// Attention_5480378269697
// MI455X (gfx1250) — matrix operand order repaired and checked by reading, not compiled, not run on hardware
//
#include <hip/hip_runtime.h>
#include <hip/hip_bf16.h>

typedef __attribute__((ext_vector_type(16))) __bf16        v16bf;
typedef __attribute__((ext_vector_type(8)))  float         v8f;
typedef __attribute__((ext_vector_type(4)))  unsigned int  v4u;
typedef __attribute__((ext_vector_type(4)))  int           v4i;

typedef __attribute__((address_space(1))) v4i* gptr_v4i;   // global int4*
typedef __attribute__((address_space(3))) v4i* lptr_v4i;   // LDS int4*

union FragAB { v16bf bf; v4u q[2]; unsigned int u[8]; };

#if __has_builtin(__builtin_amdgcn_global_load_async_to_lds_b128) && \
    __has_builtin(__builtin_amdgcn_s_wait_asynccnt)
#define USE_ASYNC_LDS 1
#else
#define USE_ASYNC_LDS 0
#endif

__device__ __forceinline__ unsigned short f32_to_bf16(float f) {
  unsigned int u = __float_as_uint(f);
  unsigned int r = u + 0x7FFFu + ((u >> 16) & 1u);   // round-to-nearest-even
  return (unsigned short)(r >> 16);
}

// ---------------------------------------------------------------------------
// elementwise f32 -> bf16
__global__ void k_f32_to_bf16(const float* __restrict__ in,
                              unsigned short* __restrict__ out, int n) {
  int i = blockIdx.x * blockDim.x + threadIdx.x;
  if (i < n) out[i] = f32_to_bf16(in[i]);
}

// 512x512 weight: WT[n*512+k] = bf16(W[k*512+n])
__global__ void k_w_transpose_bf16(const float* __restrict__ W,
                                   unsigned short* __restrict__ WT) {
  int i = blockIdx.x * blockDim.x + threadIdx.x;   // 262144 threads
  int k = i >> 9, n = i & 511;
  WT[n * 512 + k] = f32_to_bf16(W[i]);
}

// ---------------------------------------------------------------------------
// Unified bf16 WMMA GEMM:  C[z][m,n] = sum_k A[z][m,k] * BT[z][n,k]  (+ mask)
// Block = 256 threads = 8 waves arranged 2(M) x 4(N); macro tile 64(M) x 128(N).
// Each wave computes a 32x64 tile as 2x4 WMMA 16x16 subtiles (8 accumulators).
// The block stages the A(64x32) / B(128x32) K-chunk tiles in LDS once via the
// gfx1250 async-to-LDS path (double-buffered, ASYNCcnt + barriers) instead of
// per-wave global fetches (~3x less L2 traffic).
__global__ __launch_bounds__(256)
void k_gemm_wmma(const unsigned short* __restrict__ A,  long long aStride, int lda,
                 const unsigned short* __restrict__ BT, long long bStride, int ldbt,
                 const float* __restrict__ mask, long long mStride,
                 float* __restrict__ outF32,
                 unsigned short* __restrict__ outBF16, long long cStride, int ldc,
                 unsigned short* __restrict__ outBF16T, long long ldct,
                 int M, int N, int K) {
  const int lane = threadIdx.x & 31;
  const int wave = threadIdx.x >> 5;
  const int mBlk = blockIdx.y * 64;
  const int nBlk = blockIdx.x * 128;
  const int m0 = mBlk + (wave >> 2) * 32;   // wave owns 32 rows
  const int n0 = nBlk + (wave & 3) * 64;    // wave owns 64 cols
  const int z  = blockIdx.z;
  if (m0 >= M || n0 >= N) return;

  const unsigned short* Ab = A  + (long long)z * aStride;
  const unsigned short* Bb = BT + (long long)z * bStride;

  const int l15 = lane & 15;
  const int lh  = lane >> 4;

  v8f acc[2][4];
#pragma unroll
  for (int mt = 0; mt < 2; ++mt)
#pragma unroll
    for (int nt = 0; nt < 4; ++nt) acc[mt][nt] = (v8f){};

#if USE_ASYNC_LDS
  // -------- async-to-LDS staged, double-buffered --------
  __shared__ unsigned short shA[2][64 * 32];    // 8 KB
  __shared__ unsigned short shB[2][128 * 32];   // 16 KB

  // Per-thread segment assignment for the stage-in (16 B per async load):
  // A tile: 64 rows x 64 B = 256 segs (1/thread); B tile: 128 rows x 64 B = 512 segs (2/thread)
  const int t = threadIdx.x;
  const int rowA = t >> 2,      segA = t & 3;
  const int rowB0 = t >> 1,     segB0 = (t & 1) * 2;   // two consecutive segs per thread

  auto issueChunk = [&](int k0, int buf) {
    {
      const unsigned short* src = Ab + (long long)(mBlk + rowA) * lda + k0 + segA * 8;
      __builtin_amdgcn_global_load_async_to_lds_b128(
          (gptr_v4i)(v4i*)(unsigned short*)src,
          (lptr_v4i)(v4i*)&shA[buf][rowA * 32 + segA * 8], 0, 0);
    }
#pragma unroll
    for (int j = 0; j < 2; ++j) {
      const int seg = segB0 + j;
      const unsigned short* src = Bb + (long long)(nBlk + rowB0) * ldbt + k0 + seg * 8;
      __builtin_amdgcn_global_load_async_to_lds_b128(
          (gptr_v4i)(v4i*)(unsigned short*)src,
          (lptr_v4i)(v4i*)&shB[buf][rowB0 * 32 + seg * 8], 0, 0);
    }
  };

  const int nChunks = K >> 5;
  issueChunk(0, 0);
  for (int c = 0; c < nChunks; ++c) {
    const int buf = c & 1;
    if (c + 1 < nChunks) {
      issueChunk((c + 1) << 5, buf ^ 1);
      __builtin_amdgcn_s_wait_asynccnt(3);   // this wave's chunk-c loads done
    } else {
      __builtin_amdgcn_s_wait_asynccnt(0);
    }
    __syncthreads();                          // all waves' chunk-c data in LDS

    FragAB a[2], b[4];
#pragma unroll
    for (int mt = 0; mt < 2; ++mt) {
      const int r = (wave >> 2) * 32 + 16 * mt + l15;
      // A frag: contiguous elems [16*lh, +16) of the 32-elem row
      a[mt].q[0] = *(const v4u*)&shA[buf][r * 32 + 16 * lh];
      a[mt].q[1] = *(const v4u*)&shA[buf][r * 32 + 16 * lh + 8];
    }
#pragma unroll
    for (int nt = 0; nt < 4; ++nt) {
      const int r = (wave & 3) * 64 + 16 * nt + l15;
      // B frag (k = 2v + 16*lh): contiguous 32 B at elem 16*lh
      b[nt].q[0] = *(const v4u*)&shB[buf][r * 32 + 16 * lh];
      b[nt].q[1] = *(const v4u*)&shB[buf][r * 32 + 16 * lh + 8];
    }
#pragma unroll
    for (int mt = 0; mt < 2; ++mt)
#pragma unroll
      for (int nt = 0; nt < 4; ++nt)
        acc[mt][nt] = __builtin_amdgcn_wmma_f32_16x16x32_bf16(
            false, a[mt].bf, false, b[nt].bf, (short)0, acc[mt][nt], false, false);
    __syncthreads();                          // safe to overwrite buf next round
  }
#else
  // -------- direct global-load fallback (known-good round-2 form) --------
  long long aRow[2], bRow[4];
#pragma unroll
  for (int mt = 0; mt < 2; ++mt) aRow[mt] = (long long)(m0 + 16 * mt + l15) * lda;
#pragma unroll
  for (int nt = 0; nt < 4; ++nt) bRow[nt] = (long long)(n0 + 16 * nt + l15) * ldbt;

  for (int k0 = 0; k0 < K; k0 += 32) {
    FragAB a[2], b[4];
#pragma unroll
    for (int mt = 0; mt < 2; ++mt)
#pragma unroll
      for (int v = 0; v < 8; ++v) {
        const int ak = k0 + 2 * v + 16 * lh;
        a[mt].u[v] = *(const unsigned int*)(Ab + aRow[mt] + ak);
      }
#pragma unroll
    for (int nt = 0; nt < 4; ++nt)
#pragma unroll
      for (int v = 0; v < 8; ++v) {
        const int bk = k0 + 2 * v + 16 * lh;
        b[nt].u[v] = *(const unsigned int*)(Bb + bRow[nt] + bk);
      }
    if (k0 + 32 < K) {
      __builtin_prefetch(Ab + aRow[0] + k0 + 32, 0, 1);
      __builtin_prefetch(Bb + bRow[0] + k0 + 32, 0, 1);
    }
#pragma unroll
    for (int mt = 0; mt < 2; ++mt)
#pragma unroll
      for (int nt = 0; nt < 4; ++nt)
        acc[mt][nt] = __builtin_amdgcn_wmma_f32_16x16x32_bf16(
            false, a[mt].bf, false, b[nt].bf, (short)0, acc[mt][nt], false, false);
  }
#endif

  // C/D layout: element (m = mbase + v + 8*lh, n = nbase + l15) in acc[.][.][v]
#pragma unroll
  for (int mt = 0; mt < 2; ++mt)
#pragma unroll
    for (int nt = 0; nt < 4; ++nt)
#pragma unroll
      for (int v = 0; v < 8; ++v) {
        const int m = m0 + 16 * mt + v + 8 * lh;
        const int n = n0 + 16 * nt + l15;
        float val = acc[mt][nt][v];
        if (mask) val += mask[(long long)z * mStride + (long long)m * N + n];
        const long long ci = (long long)z * cStride + (long long)m * ldc + n;
        if (outF32)   outF32[ci]  = val;
        if (outBF16)  outBF16[ci] = f32_to_bf16(val);
        if (outBF16T) outBF16T[(long long)n * ldct + m] = f32_to_bf16(val);
      }
}

// ---------------------------------------------------------------------------
// softmax over the leading (num_clips = 8) axis; element index e spans G*L*L
__global__ void k_softmax8(const float* __restrict__ S,
                           unsigned short* __restrict__ P,
                           long long nStride, long long total) {
  long long e = (long long)blockIdx.x * blockDim.x + threadIdx.x;
  if (e >= total) return;
  float vals[8];
  float vmax = -3.402823466e+38f;
#pragma unroll
  for (int n = 0; n < 8; ++n) {
    float s = S[n * nStride + e];
    vals[n] = s;
    vmax = fmaxf(vmax, s);
  }
  float sum = 0.f;
#pragma unroll
  for (int n = 0; n < 8; ++n) { vals[n] = __expf(vals[n] - vmax); sum += vals[n]; }
  const float inv = 1.f / sum;
#pragma unroll
  for (int n = 0; n < 8; ++n) P[n * nStride + e] = f32_to_bf16(vals[n] * inv);
}

// ---------------------------------------------------------------------------
// y = LayerNorm(X + Y, eps=1e-3) * gamma + beta ; one 256-thread block per row
__global__ __launch_bounds__(256)
void k_add_layernorm(const float* __restrict__ X, const float* __restrict__ Y,
                     const float* __restrict__ gamma, const float* __restrict__ beta,
                     float* __restrict__ outF32, unsigned short* __restrict__ outBF16) {
  __shared__ float red[256];
  const int t = threadIdx.x;
  const long long base = (long long)blockIdx.x * 512;
  const float a0 = X[base + t]       + Y[base + t];
  const float a1 = X[base + t + 256] + Y[base + t + 256];

  red[t] = a0 + a1;
  __syncthreads();
  for (int s = 128; s > 0; s >>= 1) { if (t < s) red[t] += red[t + s]; __syncthreads(); }
  const float mean = red[0] * (1.f / 512.f);
  __syncthreads();

  const float d0 = a0 - mean, d1 = a1 - mean;
  red[t] = d0 * d0 + d1 * d1;
  __syncthreads();
  for (int s = 128; s > 0; s >>= 1) { if (t < s) red[t] += red[t + s]; __syncthreads(); }
  const float rstd = rsqrtf(red[0] * (1.f / 512.f) + 1e-3f);

  const float o0 = d0 * rstd * gamma[t]       + beta[t];
  const float o1 = d1 * rstd * gamma[t + 256] + beta[t + 256];
  if (outF32)  { outF32[base + t] = o0;               outF32[base + t + 256] = o1; }
  if (outBF16) { outBF16[base + t] = f32_to_bf16(o0); outBF16[base + t + 256] = f32_to_bf16(o1); }
}

// ---------------------------------------------------------------------------
extern "C" void kernel_launch(void* const* d_in, const int* in_sizes, int n_in,
                              void* d_out, int out_size, void* d_ws, size_t ws_size,
                              hipStream_t stream) {
  (void)in_sizes; (void)n_in; (void)out_size; (void)ws_size;
  const int NC = 8, C = 512, ROWS = 8192;
  const int BL = NC * ROWS;                 // 65536 total rows
  const long long WM = 512LL * 512LL;       // weight matrix elems

  const float* frame  = (const float*)d_in[0];
  const float* bert   = (const float*)d_in[1];
  const float* masks4[4] = { (const float*)d_in[2], (const float*)d_in[3],
                             (const float*)d_in[4], (const float*)d_in[2] };
  const float* ln1_g = (const float*)d_in[17];
  const float* ln1_b = (const float*)d_in[18];
  const float* ln2_g = (const float*)d_in[19];
  const float* ln2_b = (const float*)d_in[20];

  // ---- workspace carving ----
  char* wp = (char*)d_ws;
  auto carve = [&](size_t bytes) -> void* {
    void* r = (void*)wp;
    wp += (bytes + 255) & ~(size_t)255;
    return r;
  };
  unsigned short* WT  = (unsigned short*)carve(12 * WM * 2);        // 6 MB, 12x bf16 W^T
  unsigned short* Abf = (unsigned short*)carve((size_t)BL * C * 2); // 64 MB activation bf16
  unsigned short* Qb  = (unsigned short*)carve((size_t)BL * C * 2);
  unsigned short* Kb  = (unsigned short*)carve((size_t)BL * C * 2);
  unsigned short* VTb = (unsigned short*)carve((size_t)BL * C * 2); // V transposed: [c][row]
  unsigned short* FFb = (unsigned short*)carve((size_t)BL * C * 2);
  unsigned short* Pb  = (unsigned short*)carve((size_t)NC * 16 * 512 * 512 * 2); // 64 MB
  float* Sb = (float*)carve((size_t)NC * 16 * 512 * 512 * 4);       // 128 MB scores
  float* Xf = (float*)carve((size_t)BL * C * 4);                    // 128 MB dpa f32 out
  float* FF = (float*)carve((size_t)BL * C * 4);                    // 128 MB post-LN1

  // ---- weights -> bf16 transposed ----
  for (int i = 0; i < 12; ++i)
    k_w_transpose_bf16<<<1024, 256, 0, stream>>>((const float*)d_in[5 + i],
                                                 WT + (size_t)i * WM);
  // frame features -> bf16 (pass-1 projection input)
  k_f32_to_bf16<<<(BL * C + 255) / 256, 256, 0, stream>>>(frame, Abf, BL * C);

  auto gemm = [&](const unsigned short* A, long long aS, int lda,
                  const unsigned short* B, long long bS, int ldbt,
                  const float* mask, long long mS,
                  float* oF, unsigned short* oB, long long cS, int ldc,
                  unsigned short* oBT, long long ldct,
                  int M, int N, int K, int batches) {
    dim3 grid((unsigned)(N / 128), (unsigned)((M + 63) / 64), (unsigned)batches);
    k_gemm_wmma<<<grid, dim3(256), 0, stream>>>(A, aS, lda, B, bS, ldbt, mask, mS,
                                                oF, oB, cS, ldc, oBT, ldct, M, N, K);
  };

  const int Gs[4] = {32, 16, 16, 32};
  const int Ls[4] = {256, 512, 512, 256};

  for (int pass = 0; pass < 4; ++pass) {
    const int G = Gs[pass], L = Ls[pass];
    const int widx = (pass < 3) ? 3 * pass : 9;   // Wq index within WT block

    if (pass == 3) {
      // ff = LN(frame + x3)  -> FF (f32) + FFb (bf16); bert -> bf16 into Abf
      k_add_layernorm<<<BL, 256, 0, stream>>>(frame, Xf, ln1_g, ln1_b, FF, FFb);
      k_f32_to_bf16<<<(BL * C + 255) / 256, 256, 0, stream>>>(bert, Abf, BL * C);
    }
    const unsigned short* qin  = (pass == 3) ? FFb : Abf;
    const unsigned short* kvin = Abf;   // pass<3: activations, pass3: bert

    // projections: Q row-major bf16, K row-major bf16 (doubles as BT for QK^T),
    // V transposed bf16 (BT for P@V)
    gemm(qin,  0, C, WT + (size_t)(widx + 0) * WM, 0, C, nullptr, 0,
         nullptr, Qb, 0, C, nullptr, 0, BL, C, C, 1);
    gemm(kvin, 0, C, WT + (size_t)(widx + 1) * WM, 0, C, nullptr, 0,
         nullptr, Kb, 0, C, nullptr, 0, BL, C, C, 1);
    gemm(kvin, 0, C, WT + (size_t)(widx + 2) * WM, 0, C, nullptr, 0,
         nullptr, nullptr, 0, 0, VTb, BL, BL, C, C, 1);

    // scores: S[z][q,k] = Q K^T + mask  (z = clip*G + g, contiguous row blocks)
    gemm(Qb, (long long)L * C, C, Kb, (long long)L * C, C,
         masks4[pass], (long long)L * L,
         Sb, nullptr, (long long)L * L, L, nullptr, 0, L, L, C, NC * G);

    // softmax over the clip axis (8 values, stride G*L*L)
    const long long tot = (long long)G * L * L;
    k_softmax8<<<(unsigned)((tot + 255) / 256), 256, 0, stream>>>(Sb, Pb, tot, tot);

    // out = P @ V : bf16 into Abf (feeds next projections) or f32 into Xf
    if (pass < 2)
      gemm(Pb, (long long)L * L, L, VTb, L, BL, nullptr, 0,
           nullptr, Abf, (long long)L * C, C, nullptr, 0, L, C, L, NC * G);
    else
      gemm(Pb, (long long)L * L, L, VTb, L, BL, nullptr, 0,
           Xf, nullptr, (long long)L * C, C, nullptr, 0, L, C, L, NC * G);
  }

  // final: out = LN(ff + x4)
  k_add_layernorm<<<BL, 256, 0, stream>>>(FF, Xf, ln2_g, ln2_b, (float*)d_out, nullptr);
}
